// GIN_Model_77756087927557
// MI455X (gfx1250) — compile-verified, matrix operand order checked by reading, not run on hardware
//
#include <hip/hip_runtime.h>

#define N_NODES 100000
#define N_EDGES 300000
#define D_IN    128
#define HDIM    256
#define HDIM2   512
#define NLAYER  4
#define NGRAPH  512

typedef _Float16 v16h __attribute__((ext_vector_type(16)));
typedef _Float16 v8h  __attribute__((ext_vector_type(8)));
typedef float    v8f  __attribute__((ext_vector_type(8)));
typedef int      vsi4 __attribute__((__vector_size__(16)));   // matches builtin param

#if defined(__has_builtin)
#if __has_builtin(__builtin_amdgcn_global_load_async_to_lds_b128) && \
    __has_builtin(__builtin_amdgcn_s_wait_asynccnt)
#define HAVE_ASYNC 1
#else
#define HAVE_ASYNC 0
#endif
#else
#define HAVE_ASYNC 0
#endif

// 16B global -> LDS copy; async (ASYNCcnt) when available
__device__ __forceinline__ void async_b128(const void* g, void* l) {
#if HAVE_ASYNC
    __builtin_amdgcn_global_load_async_to_lds_b128(
        (__attribute__((address_space(1))) vsi4*)g,
        (__attribute__((address_space(3))) vsi4*)l, 0, 0);
#else
    *(uint4*)l = *(const uint4*)g;
#endif
}
__device__ __forceinline__ void async_wait0() {
#if HAVE_ASYNC
    __builtin_amdgcn_s_wait_asynccnt(0);
#endif
}

// ---------------------------------------------------------------------------
// f32 -> f16 (vectorized, n multiple of 4)
// ---------------------------------------------------------------------------
__global__ void k_cvt4(const float* __restrict__ s, _Float16* __restrict__ d, int n4) {
    int i = blockIdx.x * blockDim.x + threadIdx.x;
    if (i < n4) {
        float4 v = ((const float4*)s)[i];
        union { _Float16 h[4]; uint2 u; } pk;
        pk.h[0] = (_Float16)v.x; pk.h[1] = (_Float16)v.y;
        pk.h[2] = (_Float16)v.z; pk.h[3] = (_Float16)v.w;
        ((uint2*)d)[i] = pk.u;
    }
}

// ---------------------------------------------------------------------------
// f32 [K,NO] -> f16 transposed [NO,K]
// ---------------------------------------------------------------------------
__global__ void k_cvt_t(const float* __restrict__ s, _Float16* __restrict__ d,
                        int K, int NO) {
    int i = blockIdx.x * blockDim.x + threadIdx.x;
    if (i < K * NO) {
        int k = i / NO, n = i - k * NO;
        d[(size_t)n * K + k] = (_Float16)s[i];
    }
}

// ---------------------------------------------------------------------------
// zero fill
// ---------------------------------------------------------------------------
__global__ void k_zero(float* __restrict__ p, int n) {
    int i = blockIdx.x * blockDim.x + threadIdx.x;
    if (i < n) p[i] = 0.0f;
}

// ---------------------------------------------------------------------------
// WMMA GEMM: C[M,NO] = act(A[M,K] @ B[K,NO] + bias)
//   A: f16 row-major [M,K];  Bt: f16 row-major TRANSPOSED weights [NO,K]
//   block = 128 threads (4 waves), tile 64x64, wave tile 16x64, K step 32,
//   double-buffered LDS filled by GLOBAL_LOAD_ASYNC_TO_LDS_B128
// ---------------------------------------------------------------------------
template <bool RELU, bool OUT_F16>
__global__ __launch_bounds__(128)
void k_gemm(const _Float16* __restrict__ A,
            const _Float16* __restrict__ Bt,
            const float* __restrict__ bias,
            void* __restrict__ Cptr,
            int M, int K, int NO)
{
    __shared__ __align__(16) _Float16 sA [2][64 * 40];  // [row][k], stride 40 halves
    __shared__ __align__(16) _Float16 sBt[2][64 * 40];  // [n][k],   stride 40 halves

    const int tid  = threadIdx.x;
    const int wave = tid >> 5;
    const int lane = tid & 31;
    const int l16  = lane & 15;
    const int lhi  = lane >> 4;
    const int rowBase = blockIdx.x * 64;
    const int colBase = blockIdx.y * 64;

    v8f acc[4];
    #pragma unroll
    for (int t = 0; t < 4; ++t)
        #pragma unroll
        for (int i = 0; i < 8; ++i) acc[t][i] = 0.0f;

    // stage one 64x32 A tile + 64x32 Bt tile into buffer `buf` (4 b128/thread)
    auto stage = [&](int buf, int k0) {
        #pragma unroll
        for (int u = 0; u < 2; ++u) {
            int s  = tid + u * 128;          // 256 slots
            int r  = s >> 2, cg = s & 3;     // 64 rows x 4 groups of 8 halves
            int gr = rowBase + r; if (gr >= M) gr = M - 1;   // clamp (masked at store)
            async_b128(A + (size_t)gr * K + k0 + cg * 8, &sA[buf][r * 40 + cg * 8]);
        }
        #pragma unroll
        for (int u = 0; u < 2; ++u) {
            int s = tid + u * 128;
            int n = s >> 2, cg = s & 3;
            async_b128(Bt + (size_t)(colBase + n) * K + k0 + cg * 8,
                       &sBt[buf][n * 40 + cg * 8]);
        }
    };

    stage(0, 0);
    async_wait0();
    __syncthreads();

    int cur = 0;
    for (int k0 = 0; k0 < K; k0 += 32) {
        if (k0 + 32 < K) stage(cur ^ 1, k0 + 32);

        // A fragment: lanes 0-15 K0-7|16-23, lanes 16-31 K8-15|24-31
        int ar = wave * 16 + l16;
        int kb = lhi * 8;
        const _Float16* cA = sA[cur];
        const _Float16* cB = sBt[cur];
        v8h alo = *(const v8h*)&cA[ar * 40 + kb];
        v8h ahi = *(const v8h*)&cA[ar * 40 + kb + 16];
        v16h afrag;
        #pragma unroll
        for (int i = 0; i < 8; ++i) { afrag[i] = alo[i]; afrag[8 + i] = ahi[i]; }

        #pragma unroll
        for (int t = 0; t < 4; ++t) {
            int bn = t * 16 + l16;
            v8h blo = *(const v8h*)&cB[bn * 40 + kb];
            v8h bhi = *(const v8h*)&cB[bn * 40 + kb + 16];
            v16h bfrag;
            #pragma unroll
            for (int i = 0; i < 8; ++i) { bfrag[i] = blo[i]; bfrag[8 + i] = bhi[i]; }
            acc[t] = __builtin_amdgcn_wmma_f32_16x16x32_f16(
                false, afrag, false, bfrag, (short)0, acc[t], false, false);
        }

        async_wait0();
        __syncthreads();
        cur ^= 1;
    }

    // epilogue: bias + activation, coalesced in N
    float*    Cf = (float*)Cptr;
    _Float16* Ch = (_Float16*)Cptr;
    #pragma unroll
    for (int t = 0; t < 4; ++t) {
        int gn = colBase + t * 16 + l16;
        float bv = bias[gn];
        #pragma unroll
        for (int r = 0; r < 8; ++r) {
            int gm = rowBase + wave * 16 + r + lhi * 8;
            if (gm < M) {
                float v = acc[t][r] + bv;
                if (RELU) v = v > 0.0f ? v : 0.0f;
                if (OUT_F16) Ch[(size_t)gm * NO + gn] = (_Float16)v;
                else         Cf[(size_t)gm * NO + gn] = v;
            }
        }
    }
}

// ---------------------------------------------------------------------------
// z = (1+eps_l) * h     (fully overwrites z; no pre-zero needed)
// ---------------------------------------------------------------------------
__global__ void k_zinit(const float* __restrict__ h, const float* __restrict__ eps,
                        int l, float* __restrict__ z, int n) {
    int i = blockIdx.x * blockDim.x + threadIdx.x;
    if (i < n) z[i] = (1.0f + eps[l]) * h[i];
}

// ---------------------------------------------------------------------------
// edge scatter: z[dst] += h[src]
// ---------------------------------------------------------------------------
__global__ __launch_bounds__(64)
void k_scatter(const float* __restrict__ h, const int* __restrict__ src,
               const int* __restrict__ dst, float* __restrict__ z) {
    int e = blockIdx.x;
    int s = src[e], d = dst[e];
    int f = threadIdx.x * 4;
    float4 v = *(const float4*)(h + (size_t)s * HDIM + f);
    float* zp = z + (size_t)d * HDIM + f;
    atomicAdd(zp + 0, v.x); atomicAdd(zp + 1, v.y);
    atomicAdd(zp + 2, v.z); atomicAdd(zp + 3, v.w);
}

// ---------------------------------------------------------------------------
// per-channel sum / sum-of-squares
// ---------------------------------------------------------------------------
__global__ __launch_bounds__(256)
void k_stats(const float* __restrict__ z, float* __restrict__ sums,
             float* __restrict__ sq, int n_rows) {
    int c = threadIdx.x;
    int rows_per = (n_rows + gridDim.x - 1) / gridDim.x;
    int r0 = blockIdx.x * rows_per;
    int r1 = r0 + rows_per; if (r1 > n_rows) r1 = n_rows;
    float s = 0.f, s2 = 0.f;
    for (int r = r0; r < r1; ++r) {
        float v = z[(size_t)r * HDIM + c];
        s += v; s2 += v * v;
    }
    atomicAdd(&sums[c], s);
    atomicAdd(&sq[c], s2);
}

// ---------------------------------------------------------------------------
// BatchNorm (biased var) + relu -> h
// ---------------------------------------------------------------------------
__global__ void k_bnrelu(const float* __restrict__ z, const float* __restrict__ sums,
                         const float* __restrict__ sq, const float* __restrict__ gamma,
                         const float* __restrict__ beta, float* __restrict__ h, int n) {
    int i = blockIdx.x * blockDim.x + threadIdx.x;
    if (i >= n) return;
    int c = i & (HDIM - 1);
    const float invN = 1.0f / (float)N_NODES;
    float mean = sums[c] * invN;
    float var  = sq[c] * invN - mean * mean;
    float v = gamma[c] * (z[i] - mean) * rsqrtf(var + 1e-5f) + beta[c];
    h[i] = v > 0.0f ? v : 0.0f;
}

// ---------------------------------------------------------------------------
// graph pooling scatter
// ---------------------------------------------------------------------------
__global__ __launch_bounds__(64)
void k_pool(const float* __restrict__ h, const int* __restrict__ batch,
            float* __restrict__ pooled, float* __restrict__ counts) {
    int n = blockIdx.x;
    int g = batch[n];
    int f = threadIdx.x * 4;
    float4 v = *(const float4*)(h + (size_t)n * HDIM + f);
    float* pp = pooled + (size_t)g * HDIM + f;
    atomicAdd(pp + 0, v.x); atomicAdd(pp + 1, v.y);
    atomicAdd(pp + 2, v.z); atomicAdd(pp + 3, v.w);
    if (threadIdx.x == 0) atomicAdd(&counts[g], 1.0f);
}

// ---------------------------------------------------------------------------
// head: hid = relu(mean_pooled @ Wp1 + bp1)
// ---------------------------------------------------------------------------
__global__ __launch_bounds__(128)
void k_head1(const float* __restrict__ pooled, const float* __restrict__ counts,
             const float* __restrict__ Wp1, const float* __restrict__ bp1,
             float* __restrict__ hid) {
    int g = blockIdx.x, j = threadIdx.x;
    float cnt = counts[g];
    float inv = 1.0f / (cnt > 1.0f ? cnt : 1.0f);
    float acc = 0.f;
    for (int c = 0; c < HDIM; ++c)
        acc += pooled[(size_t)g * HDIM + c] * Wp1[(size_t)c * 128 + j];
    float v = acc * inv + bp1[j];
    hid[(size_t)g * 128 + j] = v > 0.0f ? v : 0.0f;
}

// ---------------------------------------------------------------------------
// head: out = hid @ Wp2 + bp2
// ---------------------------------------------------------------------------
__global__ __launch_bounds__(256)
void k_head2(const float* __restrict__ hid, const float* __restrict__ Wp2,
             const float* __restrict__ bp2, float* __restrict__ out) {
    int g = blockIdx.x * blockDim.x + threadIdx.x;
    if (g >= NGRAPH) return;
    float acc = 0.f;
    for (int j = 0; j < 128; ++j) acc += hid[(size_t)g * 128 + j] * Wp2[j];
    out[g] = acc + bp2[0];
}

// ---------------------------------------------------------------------------
extern "C" void kernel_launch(void* const* d_in, const int* in_sizes, int n_in,
                              void* d_out, int out_size, void* d_ws, size_t ws_size,
                              hipStream_t stream) {
    const float* x     = (const float*)d_in[0];
    const int*   ei    = (const int*)  d_in[1];
    const int*   batch = (const int*)  d_in[2];
    const float* W_in  = (const float*)d_in[3];
    const float* b_in  = (const float*)d_in[4];
    const float* eps   = (const float*)d_in[5];
    const float* W1    = (const float*)d_in[6];
    const float* b1    = (const float*)d_in[7];
    const float* W2    = (const float*)d_in[8];
    const float* b2    = (const float*)d_in[9];
    const float* gamma = (const float*)d_in[10];
    const float* beta  = (const float*)d_in[11];
    const float* Wp1   = (const float*)d_in[12];
    const float* bp1   = (const float*)d_in[13];
    const float* Wp2   = (const float*)d_in[14];
    const float* bp2   = (const float*)d_in[15];
    float* out = (float*)d_out;

    // ---- workspace carve-up (all offsets 256B aligned) ----
    char* w = (char*)d_ws;
    const size_t SZ_H = (size_t)N_NODES * HDIM * 4;          // 102,400,000
    float*    h     = (float*)(w);                           // [N,256] f32
    float*    z     = (float*)(w + SZ_H);                    // [N,256] f32 (also z2)
    _Float16* t16   = (_Float16*)(w + 2 * SZ_H);             // [N,512] f16
    _Float16* x16   = (_Float16*)(w + 3 * SZ_H);             // [N,128] f16
    _Float16* z16   = (_Float16*)(w + 3 * SZ_H + (size_t)N_NODES * D_IN * 2); // [N,256] f16
    _Float16* w16   = (_Float16*)((char*)z16 + (size_t)N_NODES * HDIM * 2);   // f16 W^T
    _Float16* WinT  = w16;                                   // [256,128]
    _Float16* W1T   = w16 + 32768;                           // L x [512,256]
    _Float16* W2T   = w16 + 32768 + 4 * HDIM * HDIM2;        // L x [256,512]
    char* w2 = (char*)(w16 + 32768 + 8 * HDIM * HDIM2);
    float* sums   = (float*)(w2);
    float* sumsq  = sums + HDIM;
    float* pooled = (float*)(w2 + 2048);
    float* counts = pooled + (size_t)NGRAPH * HDIM;
    float* hid    = counts + NGRAPH;
    (void)ws_size; (void)n_in; (void)in_sizes; (void)out_size;

    // ---- 1. weights -> f16 transposed ----
    k_cvt_t<<<(D_IN * HDIM + 255) / 256, 256, 0, stream>>>(W_in, WinT, D_IN, HDIM);
    for (int l = 0; l < NLAYER; ++l) {
        k_cvt_t<<<(HDIM * HDIM2 + 255) / 256, 256, 0, stream>>>(
            W1 + (size_t)l * HDIM * HDIM2, W1T + (size_t)l * HDIM * HDIM2, HDIM, HDIM2);
        k_cvt_t<<<(HDIM2 * HDIM + 255) / 256, 256, 0, stream>>>(
            W2 + (size_t)l * HDIM2 * HDIM, W2T + (size_t)l * HDIM2 * HDIM, HDIM2, HDIM);
    }

    const int gx = (N_NODES + 63) / 64;      // 1563
    const int nElemH = N_NODES * HDIM;       // 25.6M

    // ---- 2. input projection ----
    k_cvt4<<<(N_NODES * D_IN / 4 + 255) / 256, 256, 0, stream>>>(x, x16, N_NODES * D_IN / 4);
    k_gemm<true, false><<<dim3(gx, HDIM / 64), 128, 0, stream>>>(
        x16, WinT, b_in, (void*)h, N_NODES, D_IN, HDIM);

    // ---- 3. GIN layers ----
    for (int l = 0; l < NLAYER; ++l) {
        k_zinit<<<(nElemH + 255) / 256, 256, 0, stream>>>(h, eps, l, z, nElemH);
        k_scatter<<<N_EDGES, 64, 0, stream>>>(h, ei, ei + N_EDGES, z);
        k_cvt4<<<(nElemH / 4 + 255) / 256, 256, 0, stream>>>(z, z16, nElemH / 4);
        k_gemm<true, true><<<dim3(gx, HDIM2 / 64), 128, 0, stream>>>(
            z16, W1T + (size_t)l * HDIM * HDIM2, b1 + l * HDIM2,
            (void*)t16, N_NODES, HDIM, HDIM2);
        k_gemm<false, false><<<dim3(gx, HDIM / 64), 128, 0, stream>>>(
            t16, W2T + (size_t)l * HDIM2 * HDIM, b2 + l * HDIM,
            (void*)z, N_NODES, HDIM2, HDIM);
        k_zero<<<2, 256, 0, stream>>>(sums, 512);
        k_stats<<<256, 256, 0, stream>>>(z, sums, sumsq, N_NODES);
        k_bnrelu<<<(nElemH + 255) / 256, 256, 0, stream>>>(
            z, sums, sumsq, gamma + l * HDIM, beta + l * HDIM, h, nElemH);
    }

    // ---- 4. pooling + head ----
    k_zero<<<(NGRAPH * HDIM + NGRAPH + 255) / 256, 256, 0, stream>>>(
        pooled, NGRAPH * HDIM + NGRAPH);
    k_pool<<<N_NODES, 64, 0, stream>>>(h, batch, pooled, counts);
    k_head1<<<NGRAPH, 128, 0, stream>>>(pooled, counts, Wp1, bp1, hid);
    k_head2<<<2, 256, 0, stream>>>(hid, Wp2, bp2, out);
}
